// Actor_49203145343274
// MI455X (gfx1250) — matrix operand order repaired and checked by reading, not compiled, not run on hardware
//
#include <hip/hip_runtime.h>
#include <hip/hip_fp16.h>

// ---------------------------------------------------------------------------
// CDNA5 (gfx1250) WMMA types/helpers (wave32)
// ---------------------------------------------------------------------------
typedef __attribute__((ext_vector_type(16))) _Float16 v16h;
typedef __attribute__((ext_vector_type(8)))  _Float16 v8h;
typedef __attribute__((ext_vector_type(8)))  float    v8f;

union Vu { v16h v; v8h h[2]; };

#define WMMA_F16(a, b, c) \
  __builtin_amdgcn_wmma_f32_16x16x32_f16(false, (a), false, (b), (short)0, (c), false, false)

__device__ __forceinline__ float sigm(float x) { return 1.0f / (1.0f + __expf(-x)); }

// A-matrix 16x32 f16 fragment (ISA 7.12.2, interleaved K):
// lane L holds row (L&15); halves 0..7 = K k0+(L>=16?8:0)+j, halves 8..15 = +16.
// rsh: per-lane row is (row0 + lane%16) >> rsh (broadcasts x over the M slots).
__device__ __forceinline__ v16h frag_ld(const _Float16* base, int ld, int row0, int rsh, int k0) {
  int lane = threadIdx.x & 31;
  int r    = lane & 15;
  int hb   = (lane >> 4) << 3;  // 0 or 8
  const _Float16* p = base + (size_t)((row0 + r) >> rsh) * (size_t)ld + (size_t)(k0 + hb);
  Vu u;
  u.h[0] = *(const v8h*)p;
  u.h[1] = *(const v8h*)(p + 16);
  return u.v;
}

// B-matrix 32x16 f16 fragment (interleaved K, matches A):
// lane L holds column (L&15); halves 0..7 = K k0+(L>=16?8:0)+j, halves 8..15 = +16.
// Taken from row-major W[N][K] (WMMA consumes B = W^T).
__device__ __forceinline__ v16h frag_ld_b(const _Float16* base, int ld, int n0, int k0) {
  int lane = threadIdx.x & 31;
  int n    = lane & 15;
  int kb   = (lane >> 4) << 3;  // 0 or 8
  const _Float16* p = base + (size_t)(n0 + n) * (size_t)ld + (size_t)(k0 + kb);
  Vu u;
  u.h[0] = *(const v8h*)p;
  u.h[1] = *(const v8h*)(p + 16);
  return u.v;
}

// ---------------------------------------------------------------------------
// Problem constants
// ---------------------------------------------------------------------------
#define Bsz 1024
#define Tsz 64
#define Msz 64
#define Fsz 256

// ---------------------------------------------------------------------------
// f32 -> f16 convert with K padding to a multiple of 32
// ---------------------------------------------------------------------------
__global__ void k_cvt_pad(const float* __restrict__ src, _Float16* __restrict__ dst,
                          int N, int K, int Kp) {
  int i = blockIdx.x * blockDim.x + threadIdx.x;
  if (i >= N * Kp) return;
  int n = i / Kp, k = i - n * Kp;
  dst[i] = (k < K) ? (_Float16)src[(size_t)n * K + k] : (_Float16)0.0f;
}

__global__ void k_zero16(_Float16* p, int n) {
  int i = blockIdx.x * blockDim.x + threadIdx.x;
  if (i < n) p[i] = (_Float16)0.0f;
}
__global__ void k_zero32(float* p, int n) {
  int i = blockIdx.x * blockDim.x + threadIdx.x;
  if (i < n) p[i] = 0.0f;
}

// ---------------------------------------------------------------------------
// Fused early-fusion kernel: per 64-row block (rows = b*T + t):
//   stage input slice (f32->f16) in LDS, 4 projection GEMMs -> H[64][1024] in
//   LDS (ReLU), then fused = ReLU(H @ fus_W^T + b) -> global f16.
// Dynamic LDS: 64*512 (stage) + 64*1024 (H) halves = 192 KB (CDNA5 WGP LDS).
// ---------------------------------------------------------------------------
#define FUSION_LDS ((64 * 512 + 64 * 1024) * 2)

__global__ void k_fusion(const float* __restrict__ xv, const float* __restrict__ xe,
                         const float* __restrict__ xsc, const float* __restrict__ xp,
                         const _Float16* __restrict__ Wv, const _Float16* __restrict__ We,
                         const _Float16* __restrict__ Wsc, const _Float16* __restrict__ Wp,
                         const float* __restrict__ bv, const float* __restrict__ be,
                         const float* __restrict__ bsc, const float* __restrict__ bpo,
                         const _Float16* __restrict__ Wf, const float* __restrict__ bfu,
                         _Float16* __restrict__ fusedOut) {
  extern __shared__ __align__(16) _Float16 smem[];
  _Float16* stg = smem;             // 64 x Kp (max 512)
  _Float16* H   = smem + 64 * 512;  // 64 x 1024

  const int r0   = blockIdx.x * 64;
  const int wv_  = threadIdx.x >> 5;
  const int lane = threadIdx.x & 31;
  const int nc   = lane & 15;
  const int hb   = (lane >> 4) << 3;

  for (int p = 0; p < 4; ++p) {
    const float* src; const _Float16* W; const float* bias; int K, Kp, coff;
    if (p == 0)      { src = xv;  W = Wv;  bias = bv;  K = 128; Kp = 128; coff = 0;   }
    else if (p == 1) { src = xe;  W = We;  bias = be;  K = 512; Kp = 512; coff = 256; }
    else if (p == 2) { src = xsc; W = Wsc; bias = bsc; K = 365; Kp = 384; coff = 512; }
    else             { src = xp;  W = Wp;  bias = bpo; K = 512; Kp = 512; coff = 768; }

    // stage + convert
    for (int j = threadIdx.x; j < 64 * Kp; j += 256) {
      int r = j / Kp, k = j - r * Kp;
      float vval = (k < K) ? src[(size_t)(r0 + r) * K + k] : 0.0f;
      stg[r * Kp + k] = (_Float16)vval;
    }
    __syncthreads();

    // 64x256 tile GEMM: 4 row-tiles x 16 col-tiles
    for (int t = wv_; t < 64; t += 8) {
      int tm = t & 3, tn = t >> 2;
      int m0 = tm * 16, n0 = tn * 16;
      v8f acc = {};
      for (int k0 = 0; k0 < Kp; k0 += 32) {
        v16h a = frag_ld(stg, Kp, m0, 0, k0);
        v16h b = frag_ld_b(W, Kp, n0, k0);
        acc = WMMA_F16(a, b, acc);
      }
      float bb = bias[n0 + nc];
#pragma unroll
      for (int i = 0; i < 8; ++i) {
        float vval = fmaxf(acc[i] + bb, 0.0f);
        H[(size_t)(m0 + hb + i) * 1024 + coff + n0 + nc] = (_Float16)vval;
      }
    }
    __syncthreads();
  }

  // fusion GEMM: [64 x 1024] @ Wf^T -> [64 x 256], ReLU, to global f16
  for (int t = wv_; t < 64; t += 8) {
    int tm = t & 3, tn = t >> 2;
    int m0 = tm * 16, n0 = tn * 16;
    v8f acc = {};
    for (int k0 = 0; k0 < 1024; k0 += 32) {
      v16h a = frag_ld(H, 1024, m0, 0, k0);
      v16h b = frag_ld_b(Wf, 1024, n0, k0);
      acc = WMMA_F16(a, b, acc);
    }
    float bb = bfu[n0 + nc];
#pragma unroll
    for (int i = 0; i < 8; ++i) {
      float vval = fmaxf(acc[i] + bb, 0.0f);
      fusedOut[(size_t)(r0 + m0 + hb + i) * Fsz + n0 + nc] = (_Float16)vval;
    }
  }
}

// ---------------------------------------------------------------------------
// One LSTM step, high occupancy: grid = (B/16) x (F/16) = 1024 WGs of
// 128 threads (4 waves).  Wave w computes gate group w (i/f/g/o) for a
// 16x16 (row x col) slice: gates = x@Wih^T + h@Whh^T + b over K=512.
// Gate tile -> 4 KB LDS -> fused cell update.  h is ping-pong buffered
// (a WG reads all h columns but writes only its 16); c is sliced in-place.
// ---------------------------------------------------------------------------
__global__ void k_lstm_step(const _Float16* __restrict__ x, int xld,
                            const _Float16* __restrict__ hin, const float* __restrict__ cin,
                            const _Float16* __restrict__ Wih, const _Float16* __restrict__ Whh,
                            const float* __restrict__ bias,
                            _Float16* __restrict__ hout, float* __restrict__ cout) {
  __shared__ float g[4][16][16];
  const int rb   = blockIdx.x >> 4;
  const int nb   = blockIdx.x & 15;
  const int r0   = rb * 16;
  const int n0   = nb * 16;
  const int wv_  = threadIdx.x >> 5;  // gate 0..3 (i,f,g,o)
  const int lane = threadIdx.x & 31;
  const int nc   = lane & 15;
  const int hb   = (lane >> 4) << 3;

  const int gn0 = wv_ * 256 + n0;  // column in the 1024-wide gate matrix
  v8f acc = {};
  for (int k0 = 0; k0 < Fsz; k0 += 32) {
    v16h a = frag_ld(x, xld, r0, 0, k0);
    v16h b = frag_ld_b(Wih, Fsz, gn0, k0);
    acc = WMMA_F16(a, b, acc);
  }
  for (int k0 = 0; k0 < Fsz; k0 += 32) {
    v16h a = frag_ld(hin, Fsz, r0, 0, k0);
    v16h b = frag_ld_b(Whh, Fsz, gn0, k0);
    acc = WMMA_F16(a, b, acc);
  }
  float bb = bias[gn0 + nc];
#pragma unroll
  for (int i = 0; i < 8; ++i) g[wv_][hb + i][nc] = acc[i] + bb;
  __syncthreads();

  for (int j = threadIdx.x; j < 256; j += 128) {
    int r = j >> 4, n = j & 15;
    float ig = g[0][r][n], fg = g[1][r][n], gg = g[2][r][n], og = g[3][r][n];
    size_t idx = (size_t)(r0 + r) * Fsz + (n0 + n);
    float c = sigm(fg) * cin[idx] + sigm(ig) * tanhf(gg);
    float h = sigm(og) * tanhf(c);
    cout[idx] = c;
    hout[idx] = (_Float16)h;
  }
}

// ---------------------------------------------------------------------------
// Generic WMMA GEMM: out = act( [A0 | A1] @ W^T + bias ), N = 256.
// 64 rows / workgroup.  rsh broadcasts A rows (row >> rsh).
// act: 0 = none, 1 = relu, 2 = tanh.  out32/out16 optional.
// ---------------------------------------------------------------------------
__global__ void k_gemm(const _Float16* __restrict__ A0, int lda0, int rsh0, int K0,
                       const _Float16* __restrict__ A1, int lda1, int rsh1, int K1,
                       const _Float16* __restrict__ W, int ldw,
                       const float* __restrict__ bias, int act,
                       float* __restrict__ out32, _Float16* __restrict__ out16) {
  const int r0   = blockIdx.x * 64;
  const int wv_  = threadIdx.x >> 5;
  const int lane = threadIdx.x & 31;
  const int nc   = lane & 15;
  const int hb   = (lane >> 4) << 3;

  for (int t = wv_; t < 64; t += 8) {
    int tm = t & 3, tn = t >> 2;
    int m0 = r0 + tm * 16, n0 = tn * 16;
    v8f acc = {};
    for (int k0 = 0; k0 < K0; k0 += 32) {
      v16h a = frag_ld(A0, lda0, m0, rsh0, k0);
      v16h b = frag_ld_b(W, ldw, n0, k0);
      acc = WMMA_F16(a, b, acc);
    }
    if (A1) {
      for (int k0 = 0; k0 < K1; k0 += 32) {
        v16h a = frag_ld(A1, lda1, m0, rsh1, k0);
        v16h b = frag_ld_b(W, ldw, n0, K0 + k0);
        acc = WMMA_F16(a, b, acc);
      }
    }
    float bb = bias ? bias[n0 + nc] : 0.0f;
#pragma unroll
    for (int i = 0; i < 8; ++i) {
      float vval = acc[i] + bb;
      if (act == 1) vval = fmaxf(vval, 0.0f);
      else if (act == 2) vval = tanhf(vval);
      size_t idx = (size_t)(m0 + hb + i) * Fsz + n0 + nc;
      if (out32) out32[idx] = vval;
      if (out16) out16[idx] = (_Float16)vval;
    }
  }
}

// ---------------------------------------------------------------------------
// Attention logits: t1 = ReLU([mem | x_bcast] @ W1^T + b1) (WMMA -> LDS f32),
// then logit[row] = dot(t1, w2) + b2.  64 rows / workgroup.
// ---------------------------------------------------------------------------
__global__ void k_catlogit(const _Float16* __restrict__ A0,   // [B*M, 256] f16
                           const _Float16* __restrict__ x16,  // [B, 256] f16 (row >> 6)
                           const _Float16* __restrict__ W1,   // [256][512] f16
                           const float* __restrict__ b1,
                           const float* __restrict__ w2, const float* __restrict__ b2p,
                           float* __restrict__ logit) {
  __shared__ float t1[64 * Fsz];  // 64 KB
  const int r0   = blockIdx.x * 64;
  const int wv_  = threadIdx.x >> 5;
  const int lane = threadIdx.x & 31;
  const int nc   = lane & 15;
  const int hb   = (lane >> 4) << 3;

  for (int t = wv_; t < 64; t += 8) {
    int tm = t & 3, tn = t >> 2;
    int m0 = r0 + tm * 16, n0 = tn * 16;
    v8f acc = {};
    for (int k0 = 0; k0 < Fsz; k0 += 32) {
      v16h a = frag_ld(A0, Fsz, m0, 0, k0);
      v16h b = frag_ld_b(W1, 512, n0, k0);
      acc = WMMA_F16(a, b, acc);
    }
    for (int k0 = 0; k0 < Fsz; k0 += 32) {
      v16h a = frag_ld(x16, Fsz, m0, 6, k0);
      v16h b = frag_ld_b(W1, 512, n0, Fsz + k0);
      acc = WMMA_F16(a, b, acc);
    }
    float bb = b1[n0 + nc];
#pragma unroll
    for (int i = 0; i < 8; ++i)
      t1[(tm * 16 + hb + i) * Fsz + n0 + nc] = fmaxf(acc[i] + bb, 0.0f);
  }
  __syncthreads();

  // per-row 256-dot with w2 (8 rows per wave, wave32 shuffle reduce)
  float b2 = b2p[0];
  for (int rr = wv_ * 8; rr < wv_ * 8 + 8; ++rr) {
    float s = 0.0f;
#pragma unroll
    for (int j = 0; j < 8; ++j) {
      int n = lane + 32 * j;
      s += t1[rr * Fsz + n] * w2[n];
    }
    for (int off = 16; off > 0; off >>= 1) s += __shfl_xor(s, off, 32);
    if (lane == 0) logit[r0 + rr] = s + b2;
  }
}

// ---------------------------------------------------------------------------
// Softmax over M=64 per batch (one block of 64 threads per b)
// ---------------------------------------------------------------------------
__global__ void k_softmax64(const float* __restrict__ logit, float* __restrict__ sim) {
  __shared__ float buf[64];
  int b = blockIdx.x, m = threadIdx.x;
  float v = logit[b * 64 + m];
  buf[m] = v; __syncthreads();
  for (int s = 32; s > 0; s >>= 1) { if (m < s) buf[m] = fmaxf(buf[m], buf[m + s]); __syncthreads(); }
  float mx = buf[0]; __syncthreads();
  float e = __expf(v - mx);
  buf[m] = e; __syncthreads();
  for (int s = 32; s > 0; s >>= 1) { if (m < s) buf[m] += buf[m + s]; __syncthreads(); }
  sim[b * 64 + m] = e / buf[0];
}

// ---------------------------------------------------------------------------
// f/i/c gates (3 GEMMs sharing A fragments) fused with the memory update:
//   mem_pre = (1 - sim*sig(f)) * sub_mem + sig(i)*tanh(c)*sim   -> f16
// ---------------------------------------------------------------------------
__global__ void k_fic(const _Float16* __restrict__ sm16, const _Float16* __restrict__ x16,
                      const _Float16* __restrict__ Wf, const float* __restrict__ bf,
                      const _Float16* __restrict__ Wi, const float* __restrict__ bi,
                      const _Float16* __restrict__ Wc, const float* __restrict__ bc,
                      const float* __restrict__ sim, const float* __restrict__ sub32,
                      _Float16* __restrict__ mempre16) {
  const int r0   = blockIdx.x * 64;
  const int wv_  = threadIdx.x >> 5;
  const int lane = threadIdx.x & 31;
  const int nc   = lane & 15;
  const int hb   = (lane >> 4) << 3;

  for (int t = wv_; t < 64; t += 8) {
    int tm = t & 3, tn = t >> 2;
    int m0 = r0 + tm * 16, n0 = tn * 16;
    v8f aF = {}, aI = {}, aC = {};
    for (int k0 = 0; k0 < Fsz; k0 += 32) {
      v16h a = frag_ld(sm16, Fsz, m0, 0, k0);
      aF = WMMA_F16(a, frag_ld_b(Wf, 512, n0, k0), aF);
      aI = WMMA_F16(a, frag_ld_b(Wi, 512, n0, k0), aI);
      aC = WMMA_F16(a, frag_ld_b(Wc, 512, n0, k0), aC);
    }
    for (int k0 = 0; k0 < Fsz; k0 += 32) {
      v16h a = frag_ld(x16, Fsz, m0, 6, k0);
      aF = WMMA_F16(a, frag_ld_b(Wf, 512, n0, Fsz + k0), aF);
      aI = WMMA_F16(a, frag_ld_b(Wi, 512, n0, Fsz + k0), aI);
      aC = WMMA_F16(a, frag_ld_b(Wc, 512, n0, Fsz + k0), aC);
    }
    int col = n0 + nc;
    float bbf = bf[col], bbi = bi[col], bbc = bc[col];
#pragma unroll
    for (int i = 0; i < 8; ++i) {
      int row = m0 + hb + i;
      float fg = sigm(aF[i] + bbf);
      float ig = sigm(aI[i] + bbi);
      float cg = tanhf(aC[i] + bbc);
      float sv = sim[row];
      float sub = sub32[(size_t)row * Fsz + col];
      float vval = (1.0f - sv * fg) * sub + ig * cg * sv;
      mempre16[(size_t)row * Fsz + col] = (_Float16)vval;
    }
  }
}

// ---------------------------------------------------------------------------
// sub_vec[b,n] = sum_m mem[b,m,n] * sim2[b,m]
// ---------------------------------------------------------------------------
__global__ void k_read(const float* __restrict__ mem, const float* __restrict__ sim2,
                       float* __restrict__ sv32, _Float16* __restrict__ sv16) {
  int i = blockIdx.x * blockDim.x + threadIdx.x;
  if (i >= Bsz * Fsz) return;
  int b = i >> 8, n = i & 255;
  const float* mp = mem + (size_t)b * Msz * Fsz + n;
  const float* sp = sim2 + b * Msz;
  float s = 0.0f;
  for (int m = 0; m < Msz; ++m) s += mp[(size_t)m * Fsz] * sp[m];
  sv32[i] = s;
  sv16[i] = (_Float16)s;
}

// ---------------------------------------------------------------------------
// y[b] = sigmoid(dot(final[b], pre_W) + pre_b)
// ---------------------------------------------------------------------------
__global__ void k_pre(const float* __restrict__ fin, const float* __restrict__ w,
                      const float* __restrict__ bp, float* __restrict__ y) {
  __shared__ float buf[256];
  int b = blockIdx.x;
  buf[threadIdx.x] = fin[(size_t)b * Fsz + threadIdx.x] * w[threadIdx.x];
  __syncthreads();
  for (int s = 128; s > 0; s >>= 1) {
    if (threadIdx.x < s) buf[threadIdx.x] += buf[threadIdx.x + s];
    __syncthreads();
  }
  if (threadIdx.x == 0) y[b] = sigm(buf[0] + bp[0]);
}

// ---------------------------------------------------------------------------
// Host orchestration
// ---------------------------------------------------------------------------
extern "C" void kernel_launch(void* const* d_in, const int* in_sizes, int n_in,
                              void* d_out, int out_size, void* d_ws, size_t ws_size,
                              hipStream_t stream) {
  (void)in_sizes; (void)n_in; (void)out_size; (void)ws_size;

  const float* vggish  = (const float*)d_in[0];
  const float* emotion = (const float*)d_in[2];
  const float* scene   = (const float*)d_in[3];
  const float* pose    = (const float*)d_in[4];
  const float* submem  = (const float*)d_in[5];
  const float* vggW = (const float*)d_in[9],  *vggb = (const float*)d_in[10];
  const float* emoW = (const float*)d_in[11], *emob = (const float*)d_in[12];
  const float* scnW = (const float*)d_in[13], *scnb = (const float*)d_in[14];
  const float* posW = (const float*)d_in[15], *posb = (const float*)d_in[16];
  const float* fusW = (const float*)d_in[17], *fusb = (const float*)d_in[18];
  const float* lfWih = (const float*)d_in[19], *lfb = (const float*)d_in[20];
  const float* lfWhh = (const float*)d_in[21];
  const float* lbWih = (const float*)d_in[22], *lbb = (const float*)d_in[23];
  const float* lbWhh = (const float*)d_in[24];
  const float* tmpW = (const float*)d_in[25], *tmpb = (const float*)d_in[26];
  const float* wr1W = (const float*)d_in[27], *wr1b = (const float*)d_in[28];
  const float* wr2W = (const float*)d_in[29], *wr2b = (const float*)d_in[30];
  const float* rd1W = (const float*)d_in[31], *rd1b = (const float*)d_in[32];
  const float* rd2W = (const float*)d_in[33], *rd2b = (const float*)d_in[34];
  const float* fW = (const float*)d_in[35], *fb = (const float*)d_in[36];
  const float* iW = (const float*)d_in[37], *ib = (const float*)d_in[38];
  const float* cW = (const float*)d_in[39], *cb = (const float*)d_in[40];
  const float* sdW = (const float*)d_in[41], *sdb = (const float*)d_in[42];
  const float* finW = (const float*)d_in[43], *finb = (const float*)d_in[44];
  const float* preW = (const float*)d_in[45], *preb = (const float*)d_in[46];

  float* memOut = (float*)d_out;                     // [B*M, F]
  float* yOut   = memOut + (size_t)Bsz * Msz * Fsz;  // [B]

  // workspace layout
  char* w = (char*)d_ws; size_t off = 0;
  auto take = [&](size_t bytes) -> void* {
    void* p = w + off; off = (off + bytes + 255) & ~(size_t)255; return p;
  };
  const int BT = Bsz * Tsz, BM = Bsz * Msz;
  _Float16* wf_vgg  = (_Float16*)take((size_t)256 * 128  * 2);
  _Float16* wf_emo  = (_Float16*)take((size_t)256 * 512  * 2);
  _Float16* wf_scn  = (_Float16*)take((size_t)256 * 384  * 2);
  _Float16* wf_pos  = (_Float16*)take((size_t)256 * 512  * 2);
  _Float16* wf_fus  = (_Float16*)take((size_t)256 * 1024 * 2);
  _Float16* wf_lfih = (_Float16*)take((size_t)1024 * 256 * 2);
  _Float16* wf_lfhh = (_Float16*)take((size_t)1024 * 256 * 2);
  _Float16* wf_lbih = (_Float16*)take((size_t)1024 * 256 * 2);
  _Float16* wf_lbhh = (_Float16*)take((size_t)1024 * 256 * 2);
  _Float16* wf_tmp  = (_Float16*)take((size_t)256 * 512 * 2);
  _Float16* wf_wr1  = (_Float16*)take((size_t)256 * 512 * 2);
  _Float16* wf_rd1  = (_Float16*)take((size_t)256 * 512 * 2);
  _Float16* wf_f    = (_Float16*)take((size_t)256 * 512 * 2);
  _Float16* wf_i    = (_Float16*)take((size_t)256 * 512 * 2);
  _Float16* wf_c    = (_Float16*)take((size_t)256 * 512 * 2);
  _Float16* wf_sd   = (_Float16*)take((size_t)256 * 256 * 2);
  _Float16* wf_fin  = (_Float16*)take((size_t)256 * 256 * 2);
  _Float16* fusedX  = (_Float16*)take((size_t)BT * Fsz * 2);
  _Float16* hA      = (_Float16*)take((size_t)Bsz * Fsz * 2);
  _Float16* hB      = (_Float16*)take((size_t)Bsz * Fsz * 2);
  float*    cbuf    = (float*)   take((size_t)Bsz * Fsz * 4);
  _Float16* hbwd    = (_Float16*)take((size_t)Bsz * Fsz * 2);
  float*    cbwd    = (float*)   take((size_t)Bsz * Fsz * 4);
  _Float16* zh      = (_Float16*)take((size_t)Bsz * Fsz * 2);
  float*    zc      = (float*)   take((size_t)Bsz * Fsz * 4);
  _Float16* x16     = (_Float16*)take((size_t)Bsz * Fsz * 2);
  _Float16* sm16    = (_Float16*)take((size_t)BM * Fsz * 2);
  float*    logit1  = (float*)   take((size_t)BM * 4);
  float*    sim1    = (float*)   take((size_t)BM * 4);
  float*    logit2  = (float*)   take((size_t)BM * 4);
  float*    sim2    = (float*)   take((size_t)BM * 4);
  _Float16* mpre16  = (_Float16*)take((size_t)BM * Fsz * 2);
  _Float16* mem16   = (_Float16*)take((size_t)BM * Fsz * 2);
  float*    subv32  = (float*)   take((size_t)Bsz * Fsz * 4);
  _Float16* subv16  = (_Float16*)take((size_t)Bsz * Fsz * 2);
  float*    final32 = (float*)   take((size_t)Bsz * Fsz * 4);

  auto cvt = [&](const float* s, _Float16* d, int N, int K, int Kp) {
    int n = N * Kp;
    k_cvt_pad<<<(n + 255) / 256, 256, 0, stream>>>(s, d, N, K, Kp);
  };
  cvt(vggW, wf_vgg, 256, 128, 128);
  cvt(emoW, wf_emo, 256, 512, 512);
  cvt(scnW, wf_scn, 256, 365, 384);
  cvt(posW, wf_pos, 256, 512, 512);
  cvt(fusW, wf_fus, 256, 1024, 1024);
  cvt(lfWih, wf_lfih, 1024, 256, 256);
  cvt(lfWhh, wf_lfhh, 1024, 256, 256);
  cvt(lbWih, wf_lbih, 1024, 256, 256);
  cvt(lbWhh, wf_lbhh, 1024, 256, 256);
  cvt(tmpW, wf_tmp, 256, 512, 512);
  cvt(wr1W, wf_wr1, 256, 512, 512);
  cvt(rd1W, wf_rd1, 256, 512, 512);
  cvt(fW, wf_f, 256, 512, 512);
  cvt(iW, wf_i, 256, 512, 512);
  cvt(cW, wf_c, 256, 512, 512);
  cvt(sdW, wf_sd, 256, 256, 256);
  cvt(finW, wf_fin, 256, 256, 256);
  cvt(submem, sm16, BM, 256, 256);

  // stage 1: fused projections (192 KB dynamic LDS / workgroup)
  hipFuncSetAttribute((const void*)k_fusion,
                      hipFuncAttributeMaxDynamicSharedMemorySize, FUSION_LDS);
  k_fusion<<<BT / 64, 256, FUSION_LDS, stream>>>(
      vggish, emotion, scene, pose,
      wf_vgg, wf_emo, wf_scn, wf_pos,
      vggb, emob, scnb, posb,
      wf_fus, fusb, fusedX);

  // LSTM init + 64 sequential steps (h ping-pong, c in-place per-slice)
  k_zero16<<<(Bsz * Fsz + 255) / 256, 256, 0, stream>>>(hA, Bsz * Fsz);
  k_zero32<<<(Bsz * Fsz + 255) / 256, 256, 0, stream>>>(cbuf, Bsz * Fsz);
  k_zero16<<<(Bsz * Fsz + 255) / 256, 256, 0, stream>>>(zh, Bsz * Fsz);
  k_zero32<<<(Bsz * Fsz + 255) / 256, 256, 0, stream>>>(zc, Bsz * Fsz);
  _Float16* hfwd = hA;
  for (int t = 0; t < Tsz; ++t) {
    _Float16* hin  = (t & 1) ? hB : hA;
    _Float16* hout = (t & 1) ? hA : hB;
    k_lstm_step<<<(Bsz / 16) * (Fsz / 16), 128, 0, stream>>>(
        fusedX + (size_t)t * Fsz, Tsz * Fsz, hin, cbuf,
        wf_lfih, wf_lfhh, lfb, hout, cbuf);
    hfwd = hout;
  }
  // backward direction, single step on x[T-1] from zero state
  k_lstm_step<<<(Bsz / 16) * (Fsz / 16), 128, 0, stream>>>(
      fusedX + (size_t)(Tsz - 1) * Fsz, Tsz * Fsz, zh, zc,
      wf_lbih, wf_lbhh, lbb, hbwd, cbwd);

  // x = ReLU([h_fwd | h_bwd] @ tmp_W^T + b)
  k_gemm<<<Bsz / 64, 256, 0, stream>>>(hfwd, Fsz, 0, Fsz, hbwd, Fsz, 0, Fsz,
                                       wf_tmp, 512, tmpb, 1, nullptr, x16);

  // memory write attention
  k_catlogit<<<BM / 64, 256, 0, stream>>>(sm16, x16, wf_wr1, wr1b, wr2W, wr2b, logit1);
  k_softmax64<<<Bsz, 64, 0, stream>>>(logit1, sim1);

  // f/i/c gates + memory update -> mem_pre (f16)
  k_fic<<<BM / 64, 256, 0, stream>>>(sm16, x16, wf_f, fb, wf_i, ib, wf_c, cb,
                                     sim1, submem, mpre16);

  // mem = tanh(mem_pre @ sd_W^T + b)  -> d_out (f32) + f16 copy
  k_gemm<<<BM / 64, 256, 0, stream>>>(mpre16, Fsz, 0, Fsz, nullptr, 0, 0, 0,
                                      wf_sd, 256, sdb, 2, memOut, mem16);

  // memory read attention
  k_catlogit<<<BM / 64, 256, 0, stream>>>(mem16, x16, wf_rd1, rd1b, rd2W, rd2b, logit2);
  k_softmax64<<<Bsz, 64, 0, stream>>>(logit2, sim2);
  k_read<<<(Bsz * Fsz + 255) / 256, 256, 0, stream>>>(memOut, sim2, subv32, subv16);

  // final = tanh(sub_vec @ fin_W^T + b);  y = sigmoid(final . pre_W + b)
  k_gemm<<<Bsz / 64, 256, 0, stream>>>(subv16, Fsz, 0, Fsz, nullptr, 0, 0, 0,
                                       wf_fin, 256, finb, 2, final32, nullptr);
  k_pre<<<Bsz, 256, 0, stream>>>(final32, preW, preb, yOut);
}
